// Attention_76235669504562
// MI455X (gfx1250) — matrix operand order repaired and checked by reading, not compiled, not run on hardware
//
#include <hip/hip_runtime.h>
#include <hip/hip_bf16.h>

typedef __attribute__((ext_vector_type(16))) _Float16 v16h;
typedef __attribute__((ext_vector_type(8)))  _Float16 v8h;
typedef __attribute__((ext_vector_type(8)))  float    v8f;

#define B_    4
#define N_    1024
#define DIM_  256
#define H_    8
#define DH_   32
#define POSD_ 16
#define DP_   32
#define MTOT_ (B_*N_)   // 4096 total token rows

// ---------------------------------------------------------------------------
// f32 -> f16 conversion of x
__global__ void cvt_x_kernel(const float* __restrict__ x, _Float16* __restrict__ xh, int n) {
    int i = blockIdx.x * blockDim.x + threadIdx.x;
    if (i < n) xh[i] = (_Float16)x[i];
}

// W[k][n] (f32, 256x256) -> WT[n][k] (f16) so WMMA B-fragments are contiguous
__global__ void transpose_w_kernel(const float* __restrict__ w, _Float16* __restrict__ wt) {
    int i = blockIdx.x * blockDim.x + threadIdx.x;   // 65536
    int k = i >> 8, n = i & 255;
    wt[n * 256 + k] = (_Float16)w[k * 256 + n];
}

// ---------------------------------------------------------------------------
// C[4096,256] = X[4096,256] @ W (W given transposed as WT[n][k]), f32 acc.
// mode 0: out as [b,h,tok,dh] (Q/K layout). mode 1: out as [b,h,dh,tok] (V^T).
__global__ __launch_bounds__(256) void proj_gemm_kernel(
        const _Float16* __restrict__ X, const _Float16* __restrict__ WT,
        _Float16* __restrict__ out, int mode) {
    const int lane = threadIdx.x & 31;
    const int wave = threadIdx.x >> 5;
    const int tile = blockIdx.x * 8 + wave;     // 4096 tiles = 256 rowT x 16 colT
    const int colT = tile & 15, rowT = tile >> 4;
    const int row0 = rowT * 16, col0 = colT * 16;
    const int g  = lane >> 4;        // lane group (0: lanes 0-15, 1: lanes 16-31)
    const int ln = lane & 15;

    const _Float16* arow = X  + (size_t)(row0 + ln) * DIM_;
    const _Float16* brow = WT + (size_t)(col0 + ln) * DIM_;

    v8f acc = {};
    for (int k0 = 0; k0 < DIM_; k0 += 32) {
        v8h alo = *(const v8h*)(arow + k0 + 16 * g);
        v8h ahi = *(const v8h*)(arow + k0 + 16 * g + 8);
        v16h a, b;
        for (int e = 0; e < 8; ++e) { a[e] = alo[e]; a[8 + e] = ahi[e]; }
        b = *(const v16h*)(brow + k0 + 16 * g);
        acc = __builtin_amdgcn_wmma_f32_16x16x32_f16(false, a, false, b,
                                                     (short)0, acc, false, false);
    }
    for (int v = 0; v < 8; ++v) {
        int m = row0 + v + 8 * g;
        int c = col0 + ln;
        int bb = m >> 10, tok = m & 1023;
        int h = c >> 5, d = c & 31;
        _Float16 val = (_Float16)acc[v];
        if (mode == 0)
            out[(((size_t)(bb * H_ + h) * N_) + tok) * DH_ + d] = val;
        else
            out[(((size_t)(bb * H_ + h) * DH_) + d) * N_ + tok] = val;
    }
}

// ---------------------------------------------------------------------------
// out[4096,256] (f32) = ctx[4096,256] (f16) @ Wo + bo
__global__ __launch_bounds__(256) void out_gemm_kernel(
        const _Float16* __restrict__ ctx, const _Float16* __restrict__ WoT,
        const float* __restrict__ bo, float* __restrict__ out) {
    const int lane = threadIdx.x & 31;
    const int wave = threadIdx.x >> 5;
    const int tile = blockIdx.x * 8 + wave;
    const int colT = tile & 15, rowT = tile >> 4;
    const int row0 = rowT * 16, col0 = colT * 16;
    const int g = lane >> 4, ln = lane & 15;

    const _Float16* arow = ctx + (size_t)(row0 + ln) * DIM_;
    const _Float16* brow = WoT + (size_t)(col0 + ln) * DIM_;

    v8f acc = {};
    for (int k0 = 0; k0 < DIM_; k0 += 32) {
        v8h alo = *(const v8h*)(arow + k0 + 16 * g);
        v8h ahi = *(const v8h*)(arow + k0 + 16 * g + 8);
        v16h a, b;
        for (int e = 0; e < 8; ++e) { a[e] = alo[e]; a[8 + e] = ahi[e]; }
        b = *(const v16h*)(brow + k0 + 16 * g);
        acc = __builtin_amdgcn_wmma_f32_16x16x32_f16(false, a, false, b,
                                                     (short)0, acc, false, false);
    }
    for (int v = 0; v < 8; ++v) {
        int m = row0 + v + 8 * g;
        int c = col0 + ln;
        out[(size_t)m * DIM_ + c] = acc[v] + bo[c];
    }
}

// ---------------------------------------------------------------------------
// Positional branch (rank-1 decomposition):
//   ph[b,j,h] = (relu(pos@Wp1+bp1)@Wp2+bp2)@Wh + bh
//   w[b,h,j]  = softmax_j(-ph[b,j,h])   (i-independent!)
//   pvec[b,h,d] = sum_j w[b,h,j] * V[b,h,j,d]
__global__ __launch_bounds__(256) void pos_kernel(
        const float* __restrict__ pos,
        const float* __restrict__ Wp1, const float* __restrict__ bp1,
        const float* __restrict__ Wp2, const float* __restrict__ bp2,
        const float* __restrict__ Wh,  const float* __restrict__ bh,
        const _Float16* __restrict__ Vt, float* __restrict__ pvec) {
    __shared__ float ph[N_ * H_];   // 32 KB of the 320 KB LDS
    __shared__ float red[256];
    __shared__ float pacc[DH_];
    const int b   = blockIdx.x;
    const int tid = threadIdx.x;

    for (int j = tid; j < N_; j += 256) {
        const float* pj = pos + ((size_t)b * N_ + j) * POSD_;
        float t1[POSD_];
        for (int k = 0; k < POSD_; ++k) {
            float a = bp1[k];
            for (int i2 = 0; i2 < POSD_; ++i2) a += pj[i2] * Wp1[i2 * POSD_ + k];
            t1[k] = fmaxf(a, 0.f);
        }
        float p2[DP_];
        for (int dd = 0; dd < DP_; ++dd) {
            float a = bp2[dd];
            for (int k = 0; k < POSD_; ++k) a += t1[k] * Wp2[k * DP_ + dd];
            p2[dd] = a;
        }
        for (int hh = 0; hh < H_; ++hh) {
            float a = bh[hh];
            for (int dd = 0; dd < DP_; ++dd) a += p2[dd] * Wh[dd * H_ + hh];
            ph[j * H_ + hh] = a;
        }
    }
    __syncthreads();

    for (int hh = 0; hh < H_; ++hh) {
        float mn = __builtin_inff();
        for (int j = tid; j < N_; j += 256) mn = fminf(mn, ph[j * H_ + hh]);
        red[tid] = mn; __syncthreads();
        for (int s = 128; s > 0; s >>= 1) {
            if (tid < s) red[tid] = fminf(red[tid], red[tid + s]);
            __syncthreads();
        }
        float mnv = red[0]; __syncthreads();

        float sm = 0.f;
        for (int j = tid; j < N_; j += 256) sm += __expf(mnv - ph[j * H_ + hh]);
        red[tid] = sm; __syncthreads();
        for (int s = 128; s > 0; s >>= 1) {
            if (tid < s) red[tid] += red[tid + s];
            __syncthreads();
        }
        float ssum = red[0]; __syncthreads();

        if (tid < DH_) pacc[tid] = 0.f;
        __syncthreads();
        const _Float16* Vb = Vt + ((size_t)(b * H_ + hh)) * DH_ * N_;
        int d = tid & 31, chunk = tid >> 5;
        float part = 0.f;
        for (int j = chunk * 128; j < (chunk + 1) * 128; ++j) {
            float w = __expf(mnv - ph[j * H_ + hh]);
            part += w * (float)Vb[(size_t)d * N_ + j];
        }
        atomicAdd(&pacc[d], part);                    // ds_add_f32
        __syncthreads();
        if (tid < DH_) pvec[(size_t)(b * H_ + hh) * DH_ + tid] = pacc[tid] / ssum;
        __syncthreads();
    }
}

// ---------------------------------------------------------------------------
// Fused flash attention with gfx1250 async-to-LDS double buffering.
// Per wave: one (b,h, 16-query tile). K tile (32 tok x 32 dh = 2KB, contiguous)
// and V^T tile (32 dh x 32 tok = 2KB) staged into a private LDS slice with
// GLOBAL_LOAD_ASYNC_TO_LDS_B128 (ASYNCcnt), consumed via ds_load fragments.
__global__ __launch_bounds__(256) void attn_kernel(
        const _Float16* __restrict__ Q, const _Float16* __restrict__ K,
        const _Float16* __restrict__ Vt, const float* __restrict__ pvec,
        const float* __restrict__ gate, _Float16* __restrict__ ctx) {
    // [wave][buf][32x32 f16] = 8 * 2 * 2KB = 32KB each; 64KB/block total
    __shared__ _Float16 kst[8][2][32 * 32];
    __shared__ _Float16 vst[8][2][32 * 32];

    const int lane = threadIdx.x & 31;
    const int wave = threadIdx.x >> 5;
    const int gw   = blockIdx.x * 8 + wave;  // 2048 waves
    const int it   = gw & 63;                // 64 query tiles of 16
    const int bhid = gw >> 6;                // 32 (b,h) pairs
    const int i0   = it * 16;
    const int h    = bhid & 7, b = bhid >> 3;
    const int g    = lane >> 4, ln = lane & 15;
    const float scale = 0.17677669529663688f;   // 1/sqrt(32)

    const _Float16* Kbh = K  + (size_t)bhid * N_ * DH_;
    const _Float16* Vbh = Vt + (size_t)bhid * DH_ * N_;

    // LDS byte offsets of this wave's slices (generic ptr low 32 bits = LDS offset)
    const uint32_t kbase = (uint32_t)(uintptr_t)(&kst[wave][0][0]);
    const uint32_t vbase = (uint32_t)(uintptr_t)(&vst[wave][0][0]);

    // Issue one 2KB K tile + one 2KB V tile: 8 x B128 async loads per wave.
    auto issue_tiles = [&](int buf, int j0) {
        // K tile: contiguous 2KB at Kbh + j0*DH_
        const char* ksrc = (const char*)(Kbh + (size_t)j0 * DH_);
        for (int q = 0; q < 4; ++q) {
            uint32_t f  = (uint32_t)(q * 512 + lane * 16);
            uint32_t la = kbase + (uint32_t)buf * 2048u + f;
            uint64_t ga = (uint64_t)(uintptr_t)(ksrc + f);
            asm volatile("global_load_async_to_lds_b128 %0, %1, off"
                         :: "v"(la), "v"(ga) : "memory");
        }
        // V^T tile: 32 rows (stride N_ halves), 64B per row
        for (int q = 0; q < 4; ++q) {
            uint32_t f   = (uint32_t)(q * 512 + lane * 16);
            uint32_t row = f >> 6, byte = f & 63u;
            uint32_t la  = vbase + (uint32_t)buf * 2048u + f;
            uint64_t ga  = (uint64_t)(uintptr_t)
                ((const char*)(Vbh + (size_t)row * N_ + j0) + byte);
            asm volatile("global_load_async_to_lds_b128 %0, %1, off"
                         :: "v"(la), "v"(ga) : "memory");
        }
    };

    // Q as B-fragment of S^T = K @ Q^T: lane = col i, K-dim = d (contig 16)
    const v16h qb = *(const v16h*)(Q + ((size_t)bhid * N_ + i0 + ln) * DH_ + 16 * g);

    v8f o0 = {}, o1 = {};
    float mrun = -__builtin_inff(), lrun = 0.f;

    issue_tiles(0, 0);                       // prologue: 8 outstanding

    for (int j0 = 0; j0 < N_; j0 += 32) {
        const int buf = (j0 >> 5) & 1;
        if (j0 + 32 < N_) {
            issue_tiles(buf ^ 1, j0 + 32);   // 16 outstanding
            asm volatile("s_wait_asynccnt 8" ::: "memory");  // current buf landed
        } else {
            asm volatile("s_wait_asynccnt 0" ::: "memory");
        }

        const _Float16* ks = &kst[wave][buf][0];
        const _Float16* vs = &vst[wave][buf][0];

        // K A-fragments from LDS (rows j0+ln / j0+16+ln)
        v16h ka0, ka1;
        { v8h lo = *(const v8h*)(ks + ln * 32 + 16 * g);
          v8h hi = *(const v8h*)(ks + ln * 32 + 16 * g + 8);
          for (int e = 0; e < 8; ++e) { ka0[e] = lo[e]; ka0[8 + e] = hi[e]; } }
        { v8h lo = *(const v8h*)(ks + (16 + ln) * 32 + 16 * g);
          v8h hi = *(const v8h*)(ks + (16 + ln) * 32 + 16 * g + 8);
          for (int e = 0; e < 8; ++e) { ka1[e] = lo[e]; ka1[8 + e] = hi[e]; } }

        v8f z = {};
        v8f st0 = __builtin_amdgcn_wmma_f32_16x16x32_f16(false, ka0, false, qb,
                                                         (short)0, z, false, false);
        v8f st1 = __builtin_amdgcn_wmma_f32_16x16x32_f16(false, ka1, false, qb,
                                                         (short)0, z, false, false);

        // Lane holds S^T[j,i] for i = ln, j = 8g+{0..7} (st0), 16+8g+{0..7} (st1)
        float s[16], tmax = -__builtin_inff();
        for (int v = 0; v < 8; ++v) { s[v] = st0[v] * scale; s[8 + v] = st1[v] * scale; }
        for (int e = 0; e < 16; ++e) tmax = fmaxf(tmax, s[e]);
        tmax = fmaxf(tmax, __shfl_xor(tmax, 16, 32));
        float mnew  = fmaxf(mrun, tmax);
        float alpha = __expf(mrun - mnew);

        v16h pa;                 // P as A-fragment: exact in-lane re-layout
        float psum = 0.f;
        for (int e = 0; e < 16; ++e) {
            float pe = __expf(s[e] - mnew);
            psum += pe;
            pa[e] = (_Float16)pe;
        }
        psum += __shfl_xor(psum, 16, 32);
        lrun = lrun * alpha + psum;
        mrun = mnew;

        for (int v = 0; v < 8; ++v) {
            float al = __shfl(alpha, v + 8 * g, 32);
            o0[v] *= al; o1[v] *= al;
        }

        // V^T B-fragments from LDS: lane = col d, K-dim = token (8g+{0..7}, 16+8g+{0..7})
        v16h vb0, vb1;
        { v8h lo = *(const v8h*)(vs + ln * 32 + 8 * g);
          v8h hi = *(const v8h*)(vs + ln * 32 + 16 + 8 * g);
          for (int e = 0; e < 8; ++e) { vb0[e] = lo[e]; vb0[8 + e] = hi[e]; } }
        { v8h lo = *(const v8h*)(vs + (16 + ln) * 32 + 8 * g);
          v8h hi = *(const v8h*)(vs + (16 + ln) * 32 + 16 + 8 * g);
          for (int e = 0; e < 8; ++e) { vb1[e] = lo[e]; vb1[8 + e] = hi[e]; } }
        o0 = __builtin_amdgcn_wmma_f32_16x16x32_f16(false, pa, false, vb0,
                                                    (short)0, o0, false, false);
        o1 = __builtin_amdgcn_wmma_f32_16x16x32_f16(false, pa, false, vb1,
                                                    (short)0, o1, false, false);
    }

    float linv = 1.f / lrun;
    float gh   = 1.f / (1.f + __expf(-gate[h]));
    const float* pv = pvec + (size_t)bhid * DH_;
    for (int v = 0; v < 8; ++v) {
        float li  = __shfl(linv, v + 8 * g, 32);
        int   tok = i0 + v + 8 * g;
        float c0 = (1.f - gh) * (o0[v] * li) + gh * pv[ln];
        float c1 = (1.f - gh) * (o1[v] * li) + gh * pv[16 + ln];
        _Float16* dst = ctx + ((size_t)(b * N_ + tok)) * DIM_ + h * DH_;
        dst[ln]      = (_Float16)c0;
        dst[16 + ln] = (_Float16)c1;
    }
}

// ---------------------------------------------------------------------------
extern "C" void kernel_launch(void* const* d_in, const int* in_sizes, int n_in,
                              void* d_out, int out_size, void* d_ws, size_t ws_size,
                              hipStream_t stream) {
    (void)in_sizes; (void)n_in; (void)out_size; (void)ws_size;
    const float* x    = (const float*)d_in[0];
    // d_in[1] = deep_semantics: unused by the reference
    const float* pos  = (const float*)d_in[2];
    const float* Wq   = (const float*)d_in[3];
    const float* Wk   = (const float*)d_in[4];
    const float* Wv   = (const float*)d_in[5];
    const float* Wo   = (const float*)d_in[6];
    const float* bo   = (const float*)d_in[7];
    const float* Wp1  = (const float*)d_in[8];
    const float* bp1  = (const float*)d_in[9];
    const float* Wp2  = (const float*)d_in[10];
    const float* bp2  = (const float*)d_in[11];
    const float* Wh   = (const float*)d_in[12];
    const float* bhv  = (const float*)d_in[13];
    const float* gate = (const float*)d_in[14];
    float* out = (float*)d_out;

    char* ws = (char*)d_ws;
    size_t off = 0;
    auto alloc = [&](size_t bytes) -> void* {
        void* p = ws + off;
        off += (bytes + 255) & ~(size_t)255;
        return p;
    };
    _Float16* xh   = (_Float16*)alloc((size_t)MTOT_ * DIM_ * 2);       // 2 MB
    _Float16* WqT  = (_Float16*)alloc((size_t)DIM_ * DIM_ * 2);        // 128 KB
    _Float16* WkT  = (_Float16*)alloc((size_t)DIM_ * DIM_ * 2);
    _Float16* WvT  = (_Float16*)alloc((size_t)DIM_ * DIM_ * 2);
    _Float16* WoT  = (_Float16*)alloc((size_t)DIM_ * DIM_ * 2);
    _Float16* Qb   = (_Float16*)alloc((size_t)MTOT_ * DIM_ * 2);       // [b,h,tok,dh]
    _Float16* Kb   = (_Float16*)alloc((size_t)MTOT_ * DIM_ * 2);       // [b,h,tok,dh]
    _Float16* Vtb  = (_Float16*)alloc((size_t)MTOT_ * DIM_ * 2);       // [b,h,dh,tok]
    _Float16* ctx  = (_Float16*)alloc((size_t)MTOT_ * DIM_ * 2);       // [b,tok,dim]
    float*    pvec = (float*)alloc((size_t)B_ * H_ * DH_ * 4);         // 4 KB

    cvt_x_kernel<<<(MTOT_ * DIM_) / 256, 256, 0, stream>>>(x, xh, MTOT_ * DIM_);
    transpose_w_kernel<<<256, 256, 0, stream>>>(Wq, WqT);
    transpose_w_kernel<<<256, 256, 0, stream>>>(Wk, WkT);
    transpose_w_kernel<<<256, 256, 0, stream>>>(Wv, WvT);
    transpose_w_kernel<<<256, 256, 0, stream>>>(Wo, WoT);

    proj_gemm_kernel<<<512, 256, 0, stream>>>(xh, WqT, Qb, 0);
    proj_gemm_kernel<<<512, 256, 0, stream>>>(xh, WkT, Kb, 0);
    proj_gemm_kernel<<<512, 256, 0, stream>>>(xh, WvT, Vtb, 1);

    pos_kernel<<<B_, 256, 0, stream>>>(pos, Wp1, bp1, Wp2, bp2, Wh, bhv, Vtb, pvec);

    attn_kernel<<<256, 256, 0, stream>>>(Qb, Kb, Vtb, pvec, gate, ctx);

    out_gemm_kernel<<<512, 256, 0, stream>>>(ctx, WoT, bo, out);
}
